// DeformConv_54468775248429
// MI455X (gfx1250) — matrix operand order repaired and checked by reading, not compiled, not run on hardware
//
#include <hip/hip_runtime.h>
#include <hip/hip_bf16.h>
#include <math.h>

typedef __attribute__((ext_vector_type(16))) _Float16 v16h;
typedef __attribute__((ext_vector_type(8)))  _Float16 v8h;
typedef __attribute__((ext_vector_type(8)))  float    v8f;

#define B_    8
#define CIN   256
#define COUT  256
#define H_    64
#define W_    64
#define KK    9
#define KR    (CIN*KK)     // 2304 = reduction length
#define HW    (H_*W_)      // 4096

#define NT    16           // pixel tile (WMMA N)
#define KC    128          // channels per K-chunk
#define KS    (KC*KK)      // 1152 halves per chunk
#define LDB   (KS+8)       // padded LDS row stride (2320 B, 16B aligned)

// ---------------- kernel 1: weight f32 -> f16 A matrix [COUT][CIN*9] ----------------
__global__ void dcn_wcvt_kernel(const float* __restrict__ w, _Float16* __restrict__ aw) {
    int i = blockIdx.x * 256 + threadIdx.x;
    if (i < COUT * KR) aw[i] = (_Float16)w[i];
}

// ---------------- kernel 2: 27-channel offset conv (f32, precision-critical) --------
__global__ __launch_bounds__(256)
void dcn_offconv_kernel(const float* __restrict__ x, const float* __restrict__ wof,
                        const float* __restrict__ bof,
                        float* __restrict__ offh, float* __restrict__ offw,
                        float* __restrict__ msk) {
    int pix = blockIdx.x * 256 + threadIdx.x;      // 0..B*HW-1
    int b  = pix >> 12;
    int hw = pix & (HW - 1);
    int h  = hw >> 6, w = hw & 63;

    float acc[27];
    #pragma unroll
    for (int o = 0; o < 27; ++o) acc[o] = bof[o];

    const float* xb = x + (size_t)b * CIN * HW;
    for (int c = 0; c < CIN; ++c) {
        const float* xc = xb + c * HW;
        #pragma unroll
        for (int t = 0; t < 9; ++t) {
            int hs = h + t / 3 - 1, ws = w + t % 3 - 1;
            float xv = (hs >= 0 && hs < H_ && ws >= 0 && ws < W_) ? xc[hs * W_ + ws] : 0.f;
            const float* wp = wof + c * 9 + t;     // ((o*CIN + c)*9 + t) = o*KR + c*9 + t
            #pragma unroll
            for (int o = 0; o < 27; ++o) acc[o] = fmaf(xv, wp[o * KR], acc[o]);
        }
    }
    #pragma unroll
    for (int t = 0; t < 9; ++t) {
        int idx = (b * KK + t) * HW + hw;
        offh[idx] = acc[2 * t];
        offw[idx] = acc[2 * t + 1];
        msk[idx]  = 1.f / (1.f + expf(-acc[18 + t]));
    }
}

// ---------------- kernel 3: fused bilinear-sample + implicit GEMM (WMMA) ------------
__global__ __launch_bounds__(256)
void dcn_gemm_kernel(const float* __restrict__ x, const _Float16* __restrict__ aw,
                     const float* __restrict__ offh, const float* __restrict__ offw,
                     const float* __restrict__ msk, const float* __restrict__ bias,
                     float* __restrict__ y) {
    __shared__ _Float16 Bl[NT][LDB];       // B tile: [pixel][K-chunk] f16
    __shared__ float    t_w[NT][KK][4];    // mask-folded bilinear corner weights
    __shared__ int      t_a[NT][KK][2];    // h0, w0 corner coords

    const int tid  = threadIdx.x;
    const int tile = blockIdx.x;           // 2048 tiles = 16 pixels each
    const int b    = tile >> 8;            // 256 tiles per image
    const int hw0  = (tile & 255) << 4;    // 16 consecutive w, same row h
    const int h    = hw0 >> 6, w0 = hw0 & 63;

    // stage 0: per-(pixel,tap) sampling setup (144 threads)
    if (tid < NT * KK) {
        int p = tid / KK, t = tid % KK;
        int idx = (b * KK + t) * HW + hw0 + p;
        float ph = (float)(h + t / 3 - 1) + offh[idx];
        float pw = (float)(w0 + p + t % 3 - 1) + offw[idx];
        float m  = msk[idx];
        float fh = floorf(ph), fw = floorf(pw);
        float lh = ph - fh, lw = pw - fw;
        float hh = 1.f - lh, hw_ = 1.f - lw;
        t_w[p][t][0] = m * hh * hw_;
        t_w[p][t][1] = m * hh * lw;
        t_w[p][t][2] = m * lh * hw_;
        t_w[p][t][3] = m * lh * lw;
        t_a[p][t][0] = (int)fh;
        t_a[p][t][1] = (int)fw;
    }

    v8f c0 = {}; v8f c1 = {};
    const int lane = tid & 31;
    const int wave = tid >> 5;
    const int n    = lane & 15;            // WMMA column = pixel
    const int half = lane >> 4;            // lane half selects K sub-range
    const int m0   = wave * 2, m1 = wave * 2 + 1;   // two 16-row M tiles per wave
    const float* xb = x + (size_t)b * CIN * HW;

    for (int ck = 0; ck < 2; ++ck) {       // two K chunks of 128 channels
        __syncthreads();
        // build B tile: pixel p = tid&15, channel group tid>>4 covers 8 channels
        {
            int p = tid & 15, cg = tid >> 4;
            for (int j = 0; j < 8; ++j) {
                int cl = cg * 8 + j;                 // local channel 0..127
                const float* xc = xb + (ck * KC + cl) * HW;
                #pragma unroll
                for (int t = 0; t < KK; ++t) {
                    int h0 = t_a[p][t][0], w0i = t_a[p][t][1];
                    int h1 = h0 + 1,       w1  = w0i + 1;
                    float v00 = (h0 >= 0 && h0 < H_ && w0i >= 0 && w0i < W_) ? xc[h0 * W_ + w0i] : 0.f;
                    float v01 = (h0 >= 0 && h0 < H_ && w1  >= 0 && w1  < W_) ? xc[h0 * W_ + w1 ] : 0.f;
                    float v10 = (h1 >= 0 && h1 < H_ && w0i >= 0 && w0i < W_) ? xc[h1 * W_ + w0i] : 0.f;
                    float v11 = (h1 >= 0 && h1 < H_ && w1  >= 0 && w1  < W_) ? xc[h1 * W_ + w1 ] : 0.f;
                    float v = t_w[p][t][0] * v00 + t_w[p][t][1] * v01
                            + t_w[p][t][2] * v10 + t_w[p][t][3] * v11;
                    Bl[p][cl * KK + t] = (_Float16)v;
                }
            }
        }
        __syncthreads();

        // GEMM over this K chunk: 36 steps of K=32
        const _Float16* a0p = aw + (size_t)(m0 * 16 + n) * KR + ck * KS;
        const _Float16* a1p = aw + (size_t)(m1 * 16 + n) * KR + ck * KS;
        for (int kc = 0; kc < KS; kc += 32) {
            // B fragment: lane = col n, K runs [8h,8h+8) and [16+8h,16+8h+8)
            v16h bf;
            ((v8h*)&bf)[0] = *(const v8h*)&Bl[n][kc + 8 * half];
            ((v8h*)&bf)[1] = *(const v8h*)&Bl[n][kc + 16 + 8 * half];
            // A fragments: lane = row, K runs [8h,8h+8) and [16+8h,16+8h+8)
            v16h af0, af1;
            ((v8h*)&af0)[0] = *(const v8h*)(a0p + kc + 8 * half);
            ((v8h*)&af0)[1] = *(const v8h*)(a0p + kc + 16 + 8 * half);
            ((v8h*)&af1)[0] = *(const v8h*)(a1p + kc + 8 * half);
            ((v8h*)&af1)[1] = *(const v8h*)(a1p + kc + 16 + 8 * half);
            c0 = __builtin_amdgcn_wmma_f32_16x16x32_f16(false, af0, false, bf, (short)0, c0, false, false);
            c1 = __builtin_amdgcn_wmma_f32_16x16x32_f16(false, af1, false, bf, (short)0, c1, false, false);
        }
    }

    // store pre-BN y (+bias). C layout: VGPR r -> M = r + 8*half, N = n.
    int pix = hw0 + n;
    #pragma unroll
    for (int r = 0; r < 8; ++r) {
        int co0 = m0 * 16 + 8 * half + r;
        int co1 = m1 * 16 + 8 * half + r;
        y[((size_t)b * COUT + co0) * HW + pix] = c0[r] + bias[co0];
        y[((size_t)b * COUT + co1) * HW + pix] = c1[r] + bias[co1];
    }
}

// ---------------- kernel 4: per-channel mean / inv-std (deterministic tree) ---------
__global__ __launch_bounds__(256)
void dcn_bnstats_kernel(const float* __restrict__ y, float* __restrict__ stats) {
    int c = blockIdx.x, tid = threadIdx.x;
    float s = 0.f, ss = 0.f;
    for (int i = tid; i < B_ * HW; i += 256) {
        int b = i >> 12, hw = i & (HW - 1);
        float v = y[((size_t)b * COUT + c) * HW + hw];
        s += v; ss += v * v;
    }
    __shared__ float rs[256], rq[256];
    rs[tid] = s; rq[tid] = ss;
    __syncthreads();
    for (int k = 128; k > 0; k >>= 1) {
        if (tid < k) { rs[tid] += rs[tid + k]; rq[tid] += rq[tid + k]; }
        __syncthreads();
    }
    if (tid == 0) {
        const float inv_n = 1.f / (float)(B_ * HW);
        float mean = rs[0] * inv_n;
        float var  = rq[0] * inv_n - mean * mean;
        stats[c]        = mean;
        stats[COUT + c] = rsqrtf(var + 1e-5f);
    }
}

// ---------------- kernel 5: normalize + gamma/beta + ReLU ---------------------------
__global__ void dcn_bnapply_kernel(const float* __restrict__ y, const float* __restrict__ stats,
                                   const float* __restrict__ gamma, const float* __restrict__ beta,
                                   float* __restrict__ out) {
    int i = blockIdx.x * 256 + threadIdx.x;
    if (i >= B_ * COUT * HW) return;
    int c = (i >> 12) & (COUT - 1);
    float v = (y[i] - stats[c]) * stats[COUT + c];
    v = gamma[c] * v + beta[c];
    out[i] = v > 0.f ? v : 0.f;
}

extern "C" void kernel_launch(void* const* d_in, const int* in_sizes, int n_in,
                              void* d_out, int out_size, void* d_ws, size_t ws_size,
                              hipStream_t stream) {
    const float* x     = (const float*)d_in[0];
    const float* wof   = (const float*)d_in[1];
    const float* bof   = (const float*)d_in[2];
    const float* wgt   = (const float*)d_in[3];
    const float* bias  = (const float*)d_in[4];
    const float* gamma = (const float*)d_in[5];
    const float* beta  = (const float*)d_in[6];
    float* out = (float*)d_out;

    // workspace carve-out (all offsets 16B aligned): ~38.3 MB
    const size_t nOff = (size_t)B_ * KK * HW;          // 294912 floats each
    float*    offh  = (float*)d_ws;
    float*    offw  = offh + nOff;
    float*    msk   = offw + nOff;
    _Float16* aw    = (_Float16*)(msk + nOff);         // f16 A matrix (1.18 MB)
    float*    y     = (float*)(aw + (size_t)COUT * KR);// pre-BN output (33.5 MB)
    float*    stats = y + (size_t)B_ * COUT * HW;      // mean[256], invstd[256]

    dcn_wcvt_kernel   <<<(COUT * KR + 255) / 256, 256, 0, stream>>>(wgt, aw);
    dcn_offconv_kernel<<<(B_ * HW) / 256,        256, 0, stream>>>(x, wof, bof, offh, offw, msk);
    dcn_gemm_kernel   <<<(B_ * HW) / NT,         256, 0, stream>>>(x, aw, offh, offw, msk, bias, y);
    dcn_bnstats_kernel<<<COUT,                   256, 0, stream>>>(y, stats);
    dcn_bnapply_kernel<<<(B_ * COUT * HW + 255) / 256, 256, 0, stream>>>(y, stats, gamma, beta, out);
}
